// Mamba2_36859409334401
// MI455X (gfx1250) — matrix operand order repaired and checked by reading, not compiled, not run on hardware
//
#include <hip/hip_runtime.h>
#include <cstddef>
#include <cstdint>

#define D_MODEL   2048
#define D_STATE   128
#define D_CONV    4
#define HEADDIM   64
#define NHEADS    64
#define CHUNK     256
#define D_INNER   4096
#define CONV_DIM  4352
#define D_IN_PROJ 8512
#define BATCH     2
#define SEQ       4096
#define BL        (BATCH*SEQ)
#define NCHUNK    (SEQ/CHUNK)

typedef __attribute__((ext_vector_type(16))) __bf16 v16bf;
typedef __attribute__((ext_vector_type(4)))  __bf16 v4bf;
typedef __attribute__((ext_vector_type(8)))  float  v8f;

__device__ __forceinline__ v8f wmma_bf16(v16bf a, v16bf b, v8f c) {
  return __builtin_amdgcn_wmma_f32_16x16x32_bf16(false, a, false, b, (short)0, c,
                                                 false, false);
}

__device__ __forceinline__ v8f zero8() {
  v8f z = {0.f,0.f,0.f,0.f,0.f,0.f,0.f,0.f};
  return z;
}

__device__ __forceinline__ v4bf cvt4(float4 f) {
  v4bf r;
  r[0] = (__bf16)f.x; r[1] = (__bf16)f.y;
  r[2] = (__bf16)f.z; r[3] = (__bf16)f.w;
  return r;
}

// A-fragment: source row-major [M][ld] bf16; lane holds row m=lane&15.
// ISA 7.12.2 16-bit A 16x32: lanes0-15 K={0..7,16..23}, lanes16-31 K={8..15,24..31}.
__device__ __forceinline__ v16bf frag_a(const __bf16* base, int ld, int lane) {
  int m  = lane & 15;
  int kh = (lane & 16) ? 8 : 0;
  const __bf16* row = base + m * ld;
  v16bf f;
#pragma unroll
  for (int v = 0; v < 8; ++v) {
    int k = (v < 4) ? (kh + 2*v) : (16 + kh + 2*(v-4));
    f[2*v]   = row[k];
    f[2*v+1] = row[k+1];
  }
  return f;
}

// B-fragment: source stored as [N][ld] bf16 (N x K row-major).
// lanes0-15: K={0..7,16..23} ; lanes16-31: K={8..15,24..31}.
__device__ __forceinline__ v16bf frag_b(const __bf16* base, int ld, int lane) {
  int n    = lane & 15;
  int klow = (lane & 16) ? 8 : 0;
  const __bf16* row = base + n * ld;
  v16bf f;
#pragma unroll
  for (int v = 0; v < 8; ++v) {
    int k = (v < 4) ? (klow + 2*v) : (16 + klow + 2*(v-4));
    f[2*v]   = row[k];
    f[2*v+1] = row[k+1];
  }
  return f;
}

// ---------------------------------------------------------------------------
// GEMM: C[M,N] = A[M,K] @ Bw[K,N], f32 in/out, bf16 WMMA core.
// Block: 256 thr (8 waves), tile 256(M) x 64(N), K-step 32.
// Each wave: 2 m-tiles x 4 n-tiles = 8 WMMA/k-step; B-fragments reused.
// ---------------------------------------------------------------------------
__global__ __launch_bounds__(256) void k_gemm(const float* __restrict__ A,
                                              const float* __restrict__ Bw,
                                              float* __restrict__ C,
                                              int M, int N, int K) {
  __shared__ __bf16 As[256*32];   // 16KB
  __shared__ __bf16 Bt[64*32];    //  4KB, transposed [n][k]
  int tid  = threadIdx.x;
  int lane = tid & 31, wave = tid >> 5;
  int n0 = blockIdx.x * 64;
  int m0 = blockIdx.y * 256;
  v8f acc0[4] = {zero8(), zero8(), zero8(), zero8()};
  v8f acc1[4] = {zero8(), zero8(), zero8(), zero8()};
  for (int k0 = 0; k0 < K; k0 += 32) {
    // ---- batched vector loads into registers (loadcnt amortized) ----
    float4 ra[8];
#pragma unroll
    for (int j = 0; j < 8; ++j) {             // 256x32 A tile = 2048 float4
      int e4 = tid + j*256;
      int r = e4 >> 3, c4 = e4 & 7;
      ra[j] = *(const float4*)&A[(size_t)(m0 + r)*K + (k0 + c4*4)];
    }
    float4 rb[2];
#pragma unroll
    for (int j = 0; j < 2; ++j) {             // 32x64 B tile = 512 float4
      int e4 = tid + j*256;
      int kk = e4 >> 4, n4 = e4 & 15;
      rb[j] = *(const float4*)&Bw[(size_t)(k0 + kk)*N + (n0 + n4*4)];
    }
    // prefetch next K-tile while this one is converted/consumed
    if (k0 + 32 < K) {
      __builtin_prefetch(&A[(size_t)(m0 + tid)*K + (k0 + 32)], 0, 1);
      if (tid < 32)
        __builtin_prefetch(&Bw[(size_t)(k0 + 32 + tid)*N + n0], 0, 1);
    }
    // ---- convert + packed LDS stores ----
#pragma unroll
    for (int j = 0; j < 8; ++j) {
      int e4 = tid + j*256;
      *(v4bf*)&As[e4*4] = cvt4(ra[j]);        // ds_store_b64
    }
#pragma unroll
    for (int j = 0; j < 2; ++j) {
      int e4 = tid + j*256;
      int kk = e4 >> 4, n4 = e4 & 15;
      Bt[(n4*4 + 0)*32 + kk] = (__bf16)rb[j].x;
      Bt[(n4*4 + 1)*32 + kk] = (__bf16)rb[j].y;
      Bt[(n4*4 + 2)*32 + kk] = (__bf16)rb[j].z;
      Bt[(n4*4 + 3)*32 + kk] = (__bf16)rb[j].w;
    }
    __syncthreads();
    v16bf a0 = frag_a(As + (wave*2 + 0)*16*32, 32, lane);
    v16bf a1 = frag_a(As + (wave*2 + 1)*16*32, 32, lane);
#pragma unroll
    for (int nt = 0; nt < 4; ++nt) {
      v16bf b = frag_b(Bt + nt*16*32, 32, lane);
      acc0[nt] = wmma_bf16(a0, b, acc0[nt]);
      acc1[nt] = wmma_bf16(a1, b, acc1[nt]);
    }
    __syncthreads();
  }
  int hi8 = (lane & 16) ? 8 : 0;
#pragma unroll
  for (int nt = 0; nt < 4; ++nt) {
    int col = n0 + nt*16 + (lane & 15);
#pragma unroll
    for (int v = 0; v < 8; ++v) {
      int r0 = m0 + (wave*2 + 0)*16 + v + hi8;
      int r1 = m0 + (wave*2 + 1)*16 + v + hi8;
      C[(size_t)r0*N + col] = acc0[nt][v];
      C[(size_t)r1*N + col] = acc1[nt][v];
    }
  }
}

// ---------------------------------------------------------------------------
// Depthwise causal conv1d (K=4) + SiLU on xBC, and softplus(dt + dt_bias).
// ---------------------------------------------------------------------------
__global__ __launch_bounds__(256) void k_conv_dt(const float* __restrict__ zx,
                                                 const float* __restrict__ cw,
                                                 const float* __restrict__ cb,
                                                 const float* __restrict__ dtb,
                                                 float* __restrict__ xbc,
                                                 float* __restrict__ dts) {
  size_t idx = (size_t)blockIdx.x*256 + threadIdx.x;
  const int PER = CONV_DIM + NHEADS;
  size_t pos = idx / PER;
  int q = (int)(idx % PER);
  if (pos >= (size_t)BL) return;
  if (q < CONV_DIM) {
    int t = (int)(pos % SEQ);
    float acc = cb[q];
#pragma unroll
    for (int k = 0; k < D_CONV; ++k) {
      int tt = t - (D_CONV - 1) + k;
      if (tt >= 0)
        acc += cw[q*D_CONV + k] *
               zx[(pos - (D_CONV - 1) + k)*D_IN_PROJ + D_INNER + q];
    }
    xbc[pos*CONV_DIM + q] = acc / (1.f + __expf(-acc));   // SiLU
  } else {
    int hh = q - CONV_DIM;
    float x = zx[pos*D_IN_PROJ + D_INNER + CONV_DIM + hh] + dtb[hh];
    dts[pos*NHEADS + hh] = (x > 20.f) ? x : log1pf(__expf(x)); // softplus
  }
}

// ---------------------------------------------------------------------------
// Per (b, chunk, head): states[p,n] = sum_s exp(Acs_L - Acs_s)*dt_s*x[s,p]*B[s,n]
// WMMA GEMM M=64 (p), N=128 (n), K=256 (s). Also emits chunk decay exp(Acs_L).
// ---------------------------------------------------------------------------
__global__ __launch_bounds__(256) void k_states(const float* __restrict__ xbc,
                                                const float* __restrict__ dts,
                                                const float* __restrict__ Alog,
                                                float* __restrict__ states,
                                                float* __restrict__ cdecay) {
  extern __shared__ char smem[];
  __bf16* XwT = (__bf16*)smem;              // [64][256]  32KB
  __bf16* Bt  = (__bf16*)(smem + 32768);    // [128][256] 64KB
  float* sAcs = (float*)(smem + 98304);     // [256]
  float* sdt  = (float*)(smem + 99328);     // [256]
  int h = blockIdx.x, c = blockIdx.y, b = blockIdx.z;
  int tid = threadIdx.x;
  float Ah = -__expf(Alog[h]);
  size_t pos = (size_t)b*SEQ + (size_t)c*CHUNK + tid;
  float dtv = dts[pos*NHEADS + h];
  sdt[tid]  = dtv;
  sAcs[tid] = dtv * Ah;
  __syncthreads();
  for (int off = 1; off < 256; off <<= 1) {   // inclusive Hillis-Steele scan
    float v   = sAcs[tid];
    float add = (tid >= off) ? sAcs[tid - off] : 0.f;
    __syncthreads();
    sAcs[tid] = v + add;
    __syncthreads();
  }
  float alast = sAcs[255];
  float w = __expf(alast - sAcs[tid]) * sdt[tid];
  const float4* row4 = (const float4*)(xbc + pos*CONV_DIM);
#pragma unroll 4
  for (int p4 = 0; p4 < 16; ++p4) {           // x slice, transposed store
    float4 f = row4[h*(HEADDIM/4) + p4];
    XwT[(p4*4 + 0)*CHUNK + tid] = (__bf16)(f.x * w);
    XwT[(p4*4 + 1)*CHUNK + tid] = (__bf16)(f.y * w);
    XwT[(p4*4 + 2)*CHUNK + tid] = (__bf16)(f.z * w);
    XwT[(p4*4 + 3)*CHUNK + tid] = (__bf16)(f.w * w);
  }
#pragma unroll 4
  for (int n4 = 0; n4 < 32; ++n4) {           // B slice, transposed store
    float4 f = row4[D_INNER/4 + n4];
    Bt[(n4*4 + 0)*CHUNK + tid] = (__bf16)f.x;
    Bt[(n4*4 + 1)*CHUNK + tid] = (__bf16)f.y;
    Bt[(n4*4 + 2)*CHUNK + tid] = (__bf16)f.z;
    Bt[(n4*4 + 3)*CHUNK + tid] = (__bf16)f.w;
  }
  if (tid == 0) cdecay[((size_t)b*NCHUNK + c)*NHEADS + h] = __expf(alast);
  __syncthreads();

  int lane = tid & 31, wave = tid >> 5;
  int mt = wave & 3, ng = wave >> 2;
  v8f acc[4] = {zero8(), zero8(), zero8(), zero8()};
  for (int k0 = 0; k0 < CHUNK; k0 += 32) {
    v16bf a = frag_a(XwT + mt*16*CHUNK + k0, CHUNK, lane);
#pragma unroll
    for (int nt = 0; nt < 4; ++nt) {
      v16bf bb = frag_b(Bt + (ng*4 + nt)*16*CHUNK + k0, CHUNK, lane);
      acc[nt] = wmma_bf16(a, bb, acc[nt]);
    }
  }
  float* sp = states + (((size_t)b*NCHUNK + c)*NHEADS + h)*(HEADDIM*D_STATE);
  int hi8 = (lane & 16) ? 8 : 0;
#pragma unroll
  for (int nt = 0; nt < 4; ++nt) {
    int n = (ng*4 + nt)*16 + (lane & 15);
#pragma unroll
    for (int v = 0; v < 8; ++v) {
      int p = mt*16 + v + hi8;
      sp[p*D_STATE + n] = acc[nt][v];
    }
  }
}

// ---------------------------------------------------------------------------
// Inter-chunk recurrence: prev[c] = S; S = S*decay[c] + states[c].
// ---------------------------------------------------------------------------
__global__ __launch_bounds__(256) void k_scan(const float* __restrict__ states,
                                              const float* __restrict__ cdecay,
                                              float* __restrict__ prev) {
  int b = blockIdx.x >> 6, h = blockIdx.x & 63;
  int tid = threadIdx.x;
  float4 S[8];
#pragma unroll
  for (int i = 0; i < 8; ++i) S[i] = make_float4(0.f, 0.f, 0.f, 0.f);
  for (int c = 0; c < NCHUNK; ++c) {
    float d = cdecay[((size_t)b*NCHUNK + c)*NHEADS + h];
    size_t base4 = ((((size_t)b*NCHUNK + c)*NHEADS + h)*(HEADDIM*D_STATE)) / 4;
    const float4* st4 = (const float4*)states;
    float4* pv4 = (float4*)prev;
#pragma unroll
    for (int i = 0; i < 8; ++i) {
      size_t e4 = base4 + tid + i*256;
      pv4[e4] = S[i];
      float4 st = st4[e4];
      S[i].x = S[i].x*d + st.x;
      S[i].y = S[i].y*d + st.y;
      S[i].z = S[i].z*d + st.z;
      S[i].w = S[i].w*d + st.w;
    }
  }
}

// ---------------------------------------------------------------------------
// Per (b, chunk, head): Y = exp(Acs_l) * (C @ prev^T)          (Y_off)
//                         + [mask(exp(Acs_l - Acs_s)) .* (C@B^T)] @ (dt*x)
//                         + D_h * x
// Three WMMA stages; C A-fragments register-hoisted (reused by Y_off and
// every s-tile GEMM). ~186KB dynamic LDS (CDNA5: 320KB/WGP, 1 WG per WGP).
// ---------------------------------------------------------------------------
__global__ __launch_bounds__(256) void k_ychunk(const float* __restrict__ xbc,
                                                const float* __restrict__ dts,
                                                const float* __restrict__ Alog,
                                                const float* __restrict__ Dp,
                                                const float* __restrict__ prev,
                                                float* __restrict__ yraw) {
  extern __shared__ char smem[];
  __bf16* Cs  = (__bf16*)smem;               // [256][128]  64KB  (A source)
  __bf16* Bs  = (__bf16*)(smem + 65536);     // [256][128]  64KB  (B^T source)
  __bf16* Xt  = (__bf16*)(smem + 131072);    // [64][256]   32KB  (dt*x, N x K)
  __bf16* Pv  = (__bf16*)(smem + 163840);    // [64][128]   16KB  (prev, p x n)
  __bf16* stg = (__bf16*)(smem + 180224);    // [8][16][32]  8KB  per-wave stage
  float* sAcs = (float*)(smem + 188416);     // [256]
  float* sdt  = (float*)(smem + 189440);     // [256]
  int h = blockIdx.x, c = blockIdx.y, b = blockIdx.z;
  int tid = threadIdx.x;
  float Ah = -__expf(Alog[h]);
  size_t pos = (size_t)b*SEQ + (size_t)c*CHUNK + tid;
  float dtv = dts[pos*NHEADS + h];
  sdt[tid]  = dtv;
  sAcs[tid] = dtv * Ah;
  __syncthreads();
  for (int off = 1; off < 256; off <<= 1) {
    float v   = sAcs[tid];
    float add = (tid >= off) ? sAcs[tid - off] : 0.f;
    __syncthreads();
    sAcs[tid] = v + add;
    __syncthreads();
  }
  const float4* row4 = (const float4*)(xbc + pos*CONV_DIM);
#pragma unroll 4
  for (int n4 = 0; n4 < 32; ++n4) {           // C and B rows: contiguous stores
    float4 fc = row4[(D_INNER + D_STATE)/4 + n4];
    float4 fb = row4[D_INNER/4 + n4];
    *(v4bf*)&Cs[tid*D_STATE + n4*4] = cvt4(fc);
    *(v4bf*)&Bs[tid*D_STATE + n4*4] = cvt4(fb);
  }
#pragma unroll 4
  for (int p4 = 0; p4 < 16; ++p4) {           // dt*x, transposed store
    float4 f = row4[h*(HEADDIM/4) + p4];
    Xt[(p4*4 + 0)*CHUNK + tid] = (__bf16)(f.x * dtv);
    Xt[(p4*4 + 1)*CHUNK + tid] = (__bf16)(f.y * dtv);
    Xt[(p4*4 + 2)*CHUNK + tid] = (__bf16)(f.z * dtv);
    Xt[(p4*4 + 3)*CHUNK + tid] = (__bf16)(f.w * dtv);
  }
  const float4* pvg4 = (const float4*)
      (prev + (((size_t)b*NCHUNK + c)*NHEADS + h)*(HEADDIM*D_STATE));
#pragma unroll
  for (int j = 0; j < 8; ++j) {
    int e4 = tid + j*256;
    *(v4bf*)&Pv[e4*4] = cvt4(pvg4[e4]);
  }
  __syncthreads();

  int lane = tid & 31, wave = tid >> 5;
  int hi8 = (lane & 16) ? 8 : 0;
  float Dh = Dp[h];
  __bf16* sb = stg + wave*16*32;
  for (int mi = 0; mi < 2; ++mi) {            // each wave owns 2 m-tiles
    int mt = wave*2 + mi;
    // hoist the C-row A-fragments: reused by Y_off and every s-tile GEMM
    v16bf aC[4];
#pragma unroll
    for (int kk = 0; kk < 4; ++kk)
      aC[kk] = frag_a(Cs + mt*16*D_STATE + kk*32, D_STATE, lane);
    v8f acc[4] = {zero8(), zero8(), zero8(), zero8()};
    // ---- Y_off: C(l,n) @ prev(p,n)^T, K = D_STATE ----
#pragma unroll
    for (int kk = 0; kk < 4; ++kk) {
#pragma unroll
      for (int nt = 0; nt < 4; ++nt) {
        v16bf bb = frag_b(Pv + nt*16*D_STATE + kk*32, D_STATE, lane);
        acc[nt] = wmma_bf16(aC[kk], bb, acc[nt]);
      }
    }
#pragma unroll
    for (int v = 0; v < 8; ++v) {             // row scale exp(Acs_l)
      float sc = __expf(sAcs[mt*16 + v + hi8]);
#pragma unroll
      for (int nt = 0; nt < 4; ++nt) acc[nt][v] *= sc;
    }
    // ---- Y_diag: G = C@B^T, masked decay, then @ (dt*x) ----
    for (int sp = 0; sp*2 <= mt; ++sp) {
#pragma unroll
      for (int half = 0; half < 2; ++half) {
        int st = sp*2 + half;
        if (st <= mt) {
          v8f g = zero8();
#pragma unroll
          for (int kk = 0; kk < 4; ++kk) {
            v16bf bb = frag_b(Bs + st*16*D_STATE + kk*32, D_STATE, lane);
            g = wmma_bf16(aC[kk], bb, g);
          }
          int scol = st*16 + (lane & 15);
          float as = sAcs[scol];
#pragma unroll
          for (int v = 0; v < 8; ++v) {
            int l = mt*16 + v + hi8;
            float wgt = (scol <= l) ? __expf(sAcs[l] - as) : 0.f;
            sb[(v + hi8)*32 + half*16 + (lane & 15)] = (__bf16)(g[v] * wgt);
          }
        } else {
#pragma unroll
          for (int v = 0; v < 8; ++v)
            sb[(v + hi8)*32 + half*16 + (lane & 15)] = (__bf16)0.f;
        }
      }
      v16bf a2 = frag_a(sb, 32, lane);        // staged 16x32 masked tile
      int k0 = sp*32;
#pragma unroll
      for (int nt = 0; nt < 4; ++nt) {
        v16bf bb = frag_b(Xt + nt*16*CHUNK + k0, CHUNK, lane);
        acc[nt] = wmma_bf16(a2, bb, acc[nt]);
      }
    }
    // ---- + D*x, store ----
#pragma unroll
    for (int nt = 0; nt < 4; ++nt) {
      int pcol = nt*16 + (lane & 15);
#pragma unroll
      for (int v = 0; v < 8; ++v) {
        int l = mt*16 + v + hi8;
        size_t rp = (size_t)b*SEQ + (size_t)c*CHUNK + l;
        float xv = xbc[rp*CONV_DIM + h*HEADDIM + pcol];
        yraw[rp*D_INNER + h*HEADDIM + pcol] = acc[nt][v] + Dh*xv;
      }
    }
  }
}

// ---------------------------------------------------------------------------
// y = yraw * silu(z); RMS-normalize over D_INNER; scale by norm_weight.
// ---------------------------------------------------------------------------
__global__ __launch_bounds__(256) void k_gate_norm(const float* __restrict__ yraw,
                                                   const float* __restrict__ zx,
                                                   const float* __restrict__ nw,
                                                   float* __restrict__ yn) {
  __shared__ float red[256];
  size_t rowi = blockIdx.x;
  int tid = threadIdx.x;
  float4 vals[4];
  float ss = 0.f;
#pragma unroll
  for (int i = 0; i < 4; ++i) {
    int c4 = tid + i*256;
    float4 z4 = *(const float4*)&zx[rowi*D_IN_PROJ + c4*4];
    float4 y4 = *(const float4*)&yraw[rowi*D_INNER + c4*4];
    float4 v;
    v.x = y4.x * (z4.x / (1.f + __expf(-z4.x)));
    v.y = y4.y * (z4.y / (1.f + __expf(-z4.y)));
    v.z = y4.z * (z4.z / (1.f + __expf(-z4.z)));
    v.w = y4.w * (z4.w / (1.f + __expf(-z4.w)));
    vals[i] = v;
    ss += v.x*v.x + v.y*v.y + v.z*v.z + v.w*v.w;
  }
  red[tid] = ss;
  __syncthreads();
  for (int o = 128; o > 0; o >>= 1) {
    if (tid < o) red[tid] += red[tid + o];
    __syncthreads();
  }
  float scale = rsqrtf(red[0]/(float)D_INNER + 1e-5f);
#pragma unroll
  for (int i = 0; i < 4; ++i) {
    int c4 = tid + i*256;
    float4 w4 = *(const float4*)&nw[c4*4];
    float4 o;
    o.x = vals[i].x*scale*w4.x;
    o.y = vals[i].y*scale*w4.y;
    o.z = vals[i].z*scale*w4.z;
    o.w = vals[i].w*scale*w4.w;
    *(float4*)&yn[rowi*D_INNER + c4*4] = o;
  }
}

// ---------------------------------------------------------------------------
extern "C" void kernel_launch(void* const* d_in, const int* in_sizes, int n_in,
                              void* d_out, int out_size, void* d_ws, size_t ws_size,
                              hipStream_t stream) {
  (void)in_sizes; (void)n_in; (void)out_size; (void)ws_size;
  const float* u    = (const float*)d_in[0];
  const float* Wi   = (const float*)d_in[1];
  const float* cw   = (const float*)d_in[2];
  const float* cb   = (const float*)d_in[3];
  const float* dtb  = (const float*)d_in[4];
  const float* Alog = (const float*)d_in[5];
  const float* Dp   = (const float*)d_in[6];
  const float* nw   = (const float*)d_in[7];
  const float* Wo   = (const float*)d_in[8];
  float* out = (float*)d_out;
  char*  ws  = (char*)d_ws;

  // workspace layout (bytes)
  float* zx     = (float*)(ws + 0);              // BL*8512 f32   278,921,216
  float* xbc    = (float*)(ws + 278921216ULL);   // BL*4352 f32   142,606,336
  float* dts    = (float*)(ws + 421527552ULL);   // BL*64 f32       2,097,152
  float* states = (float*)(ws + 423624704ULL);   // 2*16*64*8192   67,108,864
  float* prev   = (float*)(ws + 490733568ULL);   // same           67,108,864
  float* cdec   = (float*)(ws + 557842432ULL);   // 2*16*64 f32         8,192
  float* yraw   = (float*)(ws + 557850624ULL);   // BL*4096 f32   134,217,728
  float* ynorm  = xbc;   // xbc dead after k_ychunk; reuse for normalized y

  // 1) in_proj GEMM: zxbcdt = u @ W_in   (8192 x 8512, K=2048)
  k_gemm<<<dim3(D_IN_PROJ/64, BL/256), 256, 0, stream>>>(u, Wi, zx,
                                                         BL, D_IN_PROJ, D_MODEL);
  // 2) causal conv + SiLU + softplus(dt)
  size_t ce = (size_t)BL*(CONV_DIM + NHEADS);
  k_conv_dt<<<(unsigned)((ce + 255)/256), 256, 0, stream>>>(zx, cw, cb, dtb,
                                                            xbc, dts);
  // 3) per-chunk states (WMMA)
  k_states<<<dim3(NHEADS, NCHUNK, BATCH), 256, 100352, stream>>>(xbc, dts, Alog,
                                                                 states, cdec);
  // 4) inter-chunk scan
  k_scan<<<BATCH*NHEADS, 256, 0, stream>>>(states, cdec, prev);
  // 5) intra-chunk Y (3x WMMA stages)
  k_ychunk<<<dim3(NHEADS, NCHUNK, BATCH), 256, 190464, stream>>>(xbc, dts, Alog,
                                                                 Dp, prev, yraw);
  // 6) gate + RMS norm
  k_gate_norm<<<BL, 256, 0, stream>>>(yraw, zx, nw, ynorm);
  // 7) out_proj GEMM -> d_out (8192 x 2048, K=4096)
  k_gemm<<<dim3(D_MODEL/64, BL/256), 256, 0, stream>>>(ynorm, Wo, out,
                                                       BL, D_MODEL, D_INNER);
}
